// BiLSTM_CRF_7679401525703
// MI455X (gfx1250) — matrix operand order repaired and checked by reading, not compiled, not run on hardware
//
#include <hip/hip_runtime.h>
#include <hip/hip_bf16.h>

// ---------------------------------------------------------------------------
// BiLSTM-CRF for MI455X (gfx1250, wave32).
//   T=4096, V=50000, E=512, H=512, NT=6
// Pipeline:
//   1) embed gather -> fp16
//   2) weight/bias pack -> fp16
//   3) WMMA GEMM: xg[T,4096] = x16[T,512] @ Wih16^T + (b_ih+b_hh)   (both dirs)
//      - B tile staged to LDS via async global->LDS (ASYNCcnt)
//   4) persistent-workgroup LSTM recurrence (one WG per direction)
//   5) output projection feats[T,6]
//   6) Viterbi DP + backtrack -> (score, path)
// ---------------------------------------------------------------------------

typedef _Float16 v16h __attribute__((ext_vector_type(16)));
typedef _Float16 v8h  __attribute__((ext_vector_type(8)));
typedef float    v8f  __attribute__((ext_vector_type(8)));
typedef int      v4i  __attribute__((ext_vector_type(4)));

// addrspace-qualified pointee typedefs for the async global->LDS builtin
typedef __attribute__((address_space(1))) v4i g_v4i;   // global (AS1)
typedef __attribute__((address_space(3))) v4i l_v4i;   // LDS (AS3)

#define TT   4096
#define EE   512
#define HH   512
#define H4   2048      // 4*H
#define NN   4096      // 2 dirs * 4H (GEMM N)
#define NTAG 6
#define NEGV (-10000.0f)

// gfx1250 async global->LDS path (ASYNCcnt). Guarded.
#if defined(__has_builtin)
#if __has_builtin(__builtin_amdgcn_global_load_async_to_lds_b128) && \
    __has_builtin(__builtin_amdgcn_s_wait_asynccnt)
#define HAVE_ASYNC_LDS 1
#endif
#endif
#ifndef HAVE_ASYNC_LDS
#define HAVE_ASYNC_LDS 0
#endif

// ---------------- fragment loaders (ISA 7.12.2 layouts) --------------------
// A (16x32 f16): lane l -> row (l&15); lanes<16 hold K {0..7,16..23},
// lanes>=16 hold K {8..15,24..31}, 8-half contiguous chunks.
__device__ __forceinline__ v16h load_fragA(const _Float16* __restrict__ p,
                                           int ld, int row0, int k0) {
  const int lane = threadIdx.x & 31;
  const int r  = row0 + (lane & 15);
  const int kb = k0 + ((lane & 16) ? 8 : 0);
  const _Float16* q = p + (size_t)r * ld + kb;
  v8h lo = *reinterpret_cast<const v8h*>(q);
  v8h hi = *reinterpret_cast<const v8h*>(q + 16);
  v16h out;
#pragma unroll
  for (int i = 0; i < 8; ++i) { out[i] = lo[i]; out[i + 8] = hi[i]; }
  return out;
}

// B (32x16 f16) fragment from an LDS tile holding 64 W-rows x 32 halves:
// lane l -> column (l&15) = W row (j*16 + l&15); lanes<16 hold K {0..7,16..23},
// lanes>=16 hold K {8..15,24..31} (same as A).
__device__ __forceinline__ v16h load_fragB_lds(const _Float16* sh, int j) {
  const int lane = threadIdx.x & 31;
  const int row = j * 16 + (lane & 15);
  const int kb  = (lane & 16) ? 8 : 0;
  const _Float16* q = sh + row * 32 + kb;
  v8h lo = *reinterpret_cast<const v8h*>(q);
  v8h hi = *reinterpret_cast<const v8h*>(q + 16);
  v16h out;
#pragma unroll
  for (int i = 0; i < 8; ++i) { out[i] = lo[i]; out[i + 8] = hi[i]; }
  return out;
}

// ---------------- small prep kernels ---------------------------------------
__global__ void embed_f16_kernel(const int* __restrict__ sent,
                                 const float* __restrict__ emb,
                                 _Float16* __restrict__ x16) {
  const int t = blockIdx.x;
  const size_t base = (size_t)sent[t] * EE;
  for (int e = threadIdx.x; e < EE; e += blockDim.x)
    x16[(size_t)t * EE + e] = (_Float16)emb[base + e];
}

__global__ void f32_to_f16_kernel(const float* __restrict__ src,
                                  _Float16* __restrict__ dst, int n) {
  int i = blockIdx.x * blockDim.x + threadIdx.x;
  if (i < n) dst[i] = (_Float16)src[i];
}

__global__ void bias_pack_kernel(const float* __restrict__ bif,
                                 const float* __restrict__ bhf,
                                 const float* __restrict__ bib,
                                 const float* __restrict__ bhb,
                                 float* __restrict__ bias) {
  int i = blockIdx.x * blockDim.x + threadIdx.x;
  if (i < H4)           bias[i] = bif[i] + bhf[i];
  else if (i < 2 * H4)  bias[i] = bib[i - H4] + bhb[i - H4];
}

// ---------------- WMMA GEMM: xg = x16 @ Wih16^T + bias ---------------------
// Block: 256 threads = 8 waves; each wave -> one 16-row M strip x 64 N cols.
// Workgroup tile: 128(M) x 64(N). B tile (64 rows x 32 halves = 4 KB) is
// staged to LDS once per k-step (async global->LDS, ASYNCcnt) and shared by
// all 8 waves. Grid: (4096/128, 4096/64) = (32, 64).
__global__ void __launch_bounds__(256)
gemm_xg_kernel(const _Float16* __restrict__ A,   // [T, E]
               const _Float16* __restrict__ W,   // [NN, E] (W^T is B)
               const float* __restrict__ bias,   // [NN]
               float* __restrict__ C) {          // [T, NN]
  const int wave = threadIdx.x >> 5;
  const int m0 = (blockIdx.x * 8 + wave) * 16;
  const int n0 = blockIdx.y * 64;

  __shared__ __align__(16) _Float16 Btile[64 * 32];  // 4 KB

  // per-thread staging slot: 4 threads per W row, 16B (8 halves) each
  const int srow = threadIdx.x >> 2;        // 0..63
  const int sch  = (threadIdx.x & 3) * 8;   // half offset within row

  v8f acc0 = {}, acc1 = {}, acc2 = {}, acc3 = {};
  for (int k0 = 0; k0 < EE; k0 += 32) {
    // ---- stage B tile into LDS ----
    {
      const _Float16* gsrc = W + (size_t)(n0 + srow) * EE + k0 + sch;
      _Float16* ldst = Btile + srow * 32 + sch;
#if HAVE_ASYNC_LDS
      __builtin_amdgcn_global_load_async_to_lds_b128(
          (g_v4i*)gsrc, (l_v4i*)ldst, 0, 0);
      __builtin_amdgcn_s_wait_asynccnt(0);
#else
      *reinterpret_cast<v8h*>(ldst) = *reinterpret_cast<const v8h*>(gsrc);
#endif
    }
    __syncthreads();

    v16h a  = load_fragA(A, EE, m0, k0);
    v16h b0 = load_fragB_lds(Btile, 0);
    v16h b1 = load_fragB_lds(Btile, 1);
    v16h b2 = load_fragB_lds(Btile, 2);
    v16h b3 = load_fragB_lds(Btile, 3);
    acc0 = __builtin_amdgcn_wmma_f32_16x16x32_f16(false, a, false, b0, (short)0, acc0, false, false);
    acc1 = __builtin_amdgcn_wmma_f32_16x16x32_f16(false, a, false, b1, (short)0, acc1, false, false);
    acc2 = __builtin_amdgcn_wmma_f32_16x16x32_f16(false, a, false, b2, (short)0, acc2, false, false);
    acc3 = __builtin_amdgcn_wmma_f32_16x16x32_f16(false, a, false, b3, (short)0, acc3, false, false);
    __syncthreads();
  }

  // C/D layout: lane -> col (lane&15); VGPR v -> row v (+8 for lanes>=16)
  const int lane  = threadIdx.x & 31;
  const int rbase = m0 + ((lane & 16) ? 8 : 0);
  v8f accs[4] = {acc0, acc1, acc2, acc3};
#pragma unroll
  for (int j = 0; j < 4; ++j) {
    const int col = n0 + j * 16 + (lane & 15);
    const float bc = bias[col];
#pragma unroll
    for (int v = 0; v < 8; ++v)
      C[(size_t)(rbase + v) * NN + col] = accs[j][v] + bc;
  }
}

// ---------------- persistent LSTM recurrence -------------------------------
__device__ __forceinline__ float sigm(float x) {
  return 1.0f / (1.0f + __expf(-x));
}

// One workgroup per direction (blockIdx.x = 0 fwd, 1 bwd), 1024 threads.
__global__ void __launch_bounds__(1024)
lstm_kernel(const float* __restrict__ xg,        // [T, NN], cols dir*2048..
            const _Float16* __restrict__ whh16,  // [2, 2048, 512]
            const float* __restrict__ h0,        // [2, 512]
            const float* __restrict__ c0,        // [2, 512]
            float* __restrict__ h_out) {         // [2, T, 512]
  const int dir = blockIdx.x;
  const int tid = threadIdx.x;

  __shared__ __align__(16) _Float16 h16[HH];
  __shared__ float cstate[HH];
  __shared__ float g[H4];

  for (int u = tid; u < HH; u += 1024) {
    h16[u]    = (_Float16)h0[dir * HH + u];
    cstate[u] = c0[dir * HH + u];
  }
  __syncthreads();

  const _Float16* Wd = whh16 + (size_t)dir * H4 * HH;
  const _Float16* w0 = Wd + (size_t)tid * HH;           // gate row tid
  const _Float16* w1 = Wd + (size_t)(tid + 1024) * HH;  // gate row tid+1024

  for (int step = 0; step < TT; ++step) {
    const int t = dir ? (TT - 1 - step) : step;
    const float* xr = xg + (size_t)t * NN + dir * H4;

    // prefetch next step's xg row (global_prefetch_b8)
    if (step + 1 < TT) {
      const int tn = dir ? (t - 1) : (t + 1);
      __builtin_prefetch(xg + (size_t)tn * NN + dir * H4 + tid, 0, 3);
    }

    // matvec: two gate rows per thread, h from LDS, W from L2 (fp16)
    float s0 = 0.0f, s1 = 0.0f;
#pragma unroll 4
    for (int k = 0; k < HH; k += 8) {
      v8h hv  = *reinterpret_cast<const v8h*>(h16 + k);
      v8h w0v = *reinterpret_cast<const v8h*>(w0 + k);
      v8h w1v = *reinterpret_cast<const v8h*>(w1 + k);
#pragma unroll
      for (int i = 0; i < 8; ++i) {
        const float hk = (float)hv[i];
        s0 = fmaf(hk, (float)w0v[i], s0);
        s1 = fmaf(hk, (float)w1v[i], s1);
      }
    }
    g[tid]        = xr[tid]        + s0;
    g[tid + 1024] = xr[tid + 1024] + s1;
    __syncthreads();

    if (tid < HH) {
      const float gi = g[tid];
      const float gf = g[HH + tid];
      const float gg = g[2 * HH + tid];
      const float go = g[3 * HH + tid];
      const float c  = sigm(gf) * cstate[tid] + sigm(gi) * tanhf(gg);
      const float h  = sigm(go) * tanhf(c);
      cstate[tid] = c;
      h16[tid]    = (_Float16)h;
      h_out[((size_t)dir * TT + t) * HH + tid] = h;
    }
    __syncthreads();
  }
}

// ---------------- output projection: feats[t][tag] -------------------------
// 192 threads = 6 warps, one tag per warp, warp-reduced dot over 1024.
__global__ void __launch_bounds__(192)
feats_kernel(const float* __restrict__ h_out,   // [2, T, 512]
             const float* __restrict__ Wout,    // [6, 1024]
             const float* __restrict__ bout,    // [6]
             float* __restrict__ feats) {       // [T, 8] (stride 8)
  const int t    = blockIdx.x;
  const int tag  = threadIdx.x >> 5;
  const int lane = threadIdx.x & 31;
  const float* hf = h_out + (size_t)t * HH;
  const float* hb = h_out + (size_t)(TT + t) * HH;
  const float* wr = Wout + tag * (2 * HH);
  float s = 0.0f;
  for (int k = lane; k < HH; k += 32)
    s += hf[k] * wr[k] + hb[k] * wr[HH + k];
#pragma unroll
  for (int off = 16; off > 0; off >>= 1) s += __shfl_down(s, off, 32);
  if (lane == 0) feats[t * 8 + tag] = s + bout[tag];
}

// ---------------- Viterbi DP + backtrack -----------------------------------
__global__ void viterbi_kernel(const float* __restrict__ feats,  // [T,8]
                               const float* __restrict__ trans,  // [6,6]
                               int* __restrict__ bps,            // [T,8]
                               float* __restrict__ out) {        // [1+T]
  const int lane = threadIdx.x;  // wave32, lanes 0..5 own tags
  float fv[NTAG];
#pragma unroll
  for (int j = 0; j < NTAG; ++j) fv[j] = (j == 4) ? 0.0f : NEGV;
  float tr[NTAG];
#pragma unroll
  for (int j = 0; j < NTAG; ++j)
    tr[j] = (lane < NTAG) ? trans[lane * NTAG + j] : 0.0f;

  for (int t = 0; t < TT; ++t) {
    float best = -3.4e38f;
    int   bp   = 0;
    if (lane < NTAG) {
#pragma unroll
      for (int j = 0; j < NTAG; ++j) {
        const float sc = fv[j] + tr[j];
        if (sc > best) { best = sc; bp = j; }
      }
      best += feats[t * 8 + lane];
      bps[t * 8 + lane] = bp;
    }
#pragma unroll
    for (int j = 0; j < NTAG; ++j) fv[j] = __shfl(best, j, 32);
  }

  if (lane == 0) {
    float bestsc = -3.4e38f;
    int   tag    = 0;
#pragma unroll
    for (int j = 0; j < NTAG; ++j) {
      const float sc = fv[j] + trans[5 * NTAG + j];  // STOP row
      if (sc > bestsc) { bestsc = sc; tag = j; }
    }
    out[0] = bestsc;
    out[TT] = (float)tag;                       // path[T-1]
    for (int t = TT - 2; t >= 0; --t) {
      tag = bps[(t + 1) * 8 + tag];
      out[1 + t] = (float)tag;
    }
  }
}

// ---------------------------------------------------------------------------
extern "C" void kernel_launch(void* const* d_in, const int* in_sizes, int n_in,
                              void* d_out, int out_size, void* d_ws, size_t ws_size,
                              hipStream_t stream) {
  const int*   sentence = (const int*)  d_in[0];
  const float* emb      = (const float*)d_in[1];
  const float* w_ih_f   = (const float*)d_in[2];
  const float* w_hh_f   = (const float*)d_in[3];
  const float* b_ih_f   = (const float*)d_in[4];
  const float* b_hh_f   = (const float*)d_in[5];
  const float* w_ih_b   = (const float*)d_in[6];
  const float* w_hh_b   = (const float*)d_in[7];
  const float* b_ih_b   = (const float*)d_in[8];
  const float* b_hh_b   = (const float*)d_in[9];
  const float* W_out    = (const float*)d_in[10];
  const float* b_out    = (const float*)d_in[11];
  const float* trans    = (const float*)d_in[12];
  const float* h0       = (const float*)d_in[13];
  const float* c0       = (const float*)d_in[14];
  float* out = (float*)d_out;

  // workspace carve-out (256B aligned)
  char*  ws  = (char*)d_ws;
  size_t off = 0;
  auto carve = [&](size_t bytes) -> void* {
    void* p = ws + off;
    off = (off + bytes + 255) & ~(size_t)255;
    return p;
  };
  _Float16* x16    = (_Float16*)carve((size_t)TT * EE * 2);        // 4 MB
  _Float16* wih16  = (_Float16*)carve((size_t)NN * EE * 2);        // 4 MB
  _Float16* whh16  = (_Float16*)carve((size_t)2 * H4 * HH * 2);    // 4 MB
  float*    bias   = (float*)   carve((size_t)NN * 4);             // 16 KB
  float*    xg     = (float*)   carve((size_t)TT * NN * 4);        // 64 MB
  float*    h_outb = (float*)   carve((size_t)2 * TT * HH * 4);    // 16 MB
  float*    feats  = (float*)   carve((size_t)TT * 8 * 4);         // 128 KB
  int*      bps    = (int*)     carve((size_t)TT * 8 * 4);         // 128 KB
  (void)ws_size; (void)in_sizes; (void)n_in; (void)out_size;

  const int nw = H4 * EE;  // 2048*512 elements per weight matrix
  f32_to_f16_kernel<<<(nw + 255) / 256, 256, 0, stream>>>(w_ih_f, wih16,       nw);
  f32_to_f16_kernel<<<(nw + 255) / 256, 256, 0, stream>>>(w_ih_b, wih16 + nw,  nw);
  f32_to_f16_kernel<<<(nw + 255) / 256, 256, 0, stream>>>(w_hh_f, whh16,       nw);
  f32_to_f16_kernel<<<(nw + 255) / 256, 256, 0, stream>>>(w_hh_b, whh16 + nw,  nw);
  bias_pack_kernel<<<(NN + 255) / 256, 256, 0, stream>>>(b_ih_f, b_hh_f, b_ih_b, b_hh_b, bias);
  embed_f16_kernel<<<TT, 256, 0, stream>>>(sentence, emb, x16);

  dim3 ggrid(TT / 128, NN / 64);
  gemm_xg_kernel<<<ggrid, 256, 0, stream>>>(x16, wih16, bias, xg);

  lstm_kernel<<<2, 1024, 0, stream>>>(xg, whh16, h0, c0, h_outb);

  feats_kernel<<<TT, 192, 0, stream>>>(h_outb, W_out, b_out, feats);

  viterbi_kernel<<<1, 32, 0, stream>>>(feats, trans, bps, out);
}
